// OccMamba_Head_16750372455153
// MI455X (gfx1250) — matrix operand order repaired and checked by reading, not compiled, not run on hardware
//
#include <hip/hip_runtime.h>

// ---------------------------------------------------------------------------
// Types for CDNA5 (gfx1250) WMMA / TDM
// ---------------------------------------------------------------------------
typedef __bf16 bf16_t;
typedef __attribute__((ext_vector_type(16))) __bf16 v16bf;
typedef __attribute__((ext_vector_type(8)))  float  v8f;
typedef __attribute__((ext_vector_type(4)))  unsigned int v4u;
typedef __attribute__((ext_vector_type(4)))  int v4i;
typedef __attribute__((ext_vector_type(8)))  int v8i;

#define TILE_M 128
#define GRID(n, b) (((n) + (b) - 1) / (b))

union FragBF { v16bf v; uint4 q[2]; };

__device__ __forceinline__ float act_apply(float v, int act) {
  if (act == 1) return v > 0.f ? v : 0.f;                       // relu
  if (act == 2) return v / (1.f + __expf(-v));                  // silu
  if (act == 3) return (v > 20.f) ? v : log1pf(__expf(v));      // softplus
  return v;
}

// ---------------------------------------------------------------------------
// WMMA GEMM:  C[M,N] = act( A[M,K16] * Wt[N16,K16]^T + bias [+ C if accum] )
// A: bf16 row-major, stride K16 (K16 % 32 == 0)
// Wt: bf16 row-major [N16,K16] (weight stored as [out,in] -> contiguous-K reads)
// C: f32, leading dim ldc. One wave owns a 16-row panel; block = 8 waves = 128 rows.
// The A panel is staged into LDS by the Tensor Data Mover; tensor_dim1 makes the
// hardware zero-fill out-of-range rows so EXEC stays all-ones through the WMMAs.
// ---------------------------------------------------------------------------
__global__ void __launch_bounds__(256)
k_gemm_wmma(const bf16_t* __restrict__ A, const bf16_t* __restrict__ Wt,
            float* __restrict__ C, const float* __restrict__ bias,
            int M, int N, int K16, int N16, int ldc, int act, int accum)
{
  extern __shared__ __align__(16) char smem_raw[];
  bf16_t* sA = reinterpret_cast<bf16_t*>(smem_raw);

  const int lane = threadIdx.x & 31;
  const int wave = threadIdx.x >> 5;
  const int m0   = blockIdx.x * TILE_M;

#if __has_builtin(__builtin_amdgcn_tensor_load_to_lds)
  if (wave == 0) {
    // ---- Tensor DMA descriptor (CDNA5 ISA): 2-D tile load ----
    unsigned long long ga = (unsigned long long)(size_t)(A + (size_t)m0 * K16);
    unsigned lds_off = (unsigned)(unsigned long long)(size_t)sA; // flat addr[31:0] == LDS offset
    int rows_rem = M - m0; if (rows_rem < 0) rows_rem = 0;

    v4u g0;
    g0[0] = 1u;                                            // count=1 (valid), user mode
    g0[1] = lds_off;                                       // lds_addr
    g0[2] = (unsigned)(ga & 0xFFFFFFFFu);                  // global_addr[31:0]
    g0[3] = (unsigned)((ga >> 32) & 0x01FFFFFFu) | (2u << 30); // addr[56:32] | type=2

    v8i g1;
    g1[0] = (int)(1u << 16);                               // wg_mask=0, data_size=1 (2 bytes)
    g1[1] = (int)(((unsigned)K16 & 0xFFFFu) << 16);        // tensor_dim0[15:0] @ bits 48..63
    g1[2] = (int)((((unsigned)K16 >> 16) & 0xFFFFu) |
                  (((unsigned)rows_rem & 0xFFFFu) << 16)); // tensor_dim0 hi | tensor_dim1 lo
    g1[3] = (int)((((unsigned)rows_rem >> 16) & 0xFFFFu) |
                  (((unsigned)K16 & 0xFFFFu) << 16));      // tensor_dim1 hi | tile_dim0=K16
    g1[4] = (int)(TILE_M & 0xFFFF);                        // tile_dim1=128, tile_dim2=0
    g1[5] = K16;                                           // tensor_dim0_stride lo32
    g1[6] = 0;                                             // stride hi | dim1_stride lo
    g1[7] = 0;

    v4i gz = {0, 0, 0, 0};
#if defined(__clang_major__) && __clang_major__ >= 23
    v8i gz8 = {0, 0, 0, 0, 0, 0, 0, 0};
    __builtin_amdgcn_tensor_load_to_lds(g0, g1, gz, gz, gz8, 0);
#else
    __builtin_amdgcn_tensor_load_to_lds(g0, g1, gz, gz, 0);
#endif
    __builtin_amdgcn_s_wait_tensorcnt(0);
  }
#else
  // Fallback: cooperative staging, zero-fill OOB rows.
  {
    int vecs = TILE_M * K16 / 8;
    for (int t = threadIdx.x; t < vecs; t += 256) {
      int row = (t * 8) / K16;
      int col = (t * 8) % K16;
      uint4 v = {0, 0, 0, 0};
      if (m0 + row < M)
        v = *reinterpret_cast<const uint4*>(A + (size_t)(m0 + row) * K16 + col);
      *reinterpret_cast<uint4*>(sA + (size_t)row * K16 + col) = v;
    }
  }
#endif
  __syncthreads();

  const int half = lane >> 4;   // 0 / 1
  const int r15  = lane & 15;
  const bf16_t* aRow = sA + (size_t)(wave * 16 + r15) * K16;

  for (int n0 = 0; n0 < N16; n0 += 16) {
    const bf16_t* bRow = Wt + (size_t)(n0 + r15) * K16;
    v8f acc = {0.f, 0.f, 0.f, 0.f, 0.f, 0.f, 0.f, 0.f};
    for (int k0 = 0; k0 < K16; k0 += 32) {
      FragBF fa, fb;
      // A 16x32 bf16 frag (ISA §7.12.2): lane-half splits K 0-7/8-15 and 16-23/24-31
      fa.q[0] = *reinterpret_cast<const uint4*>(aRow + k0 + half * 8);
      fa.q[1] = *reinterpret_cast<const uint4*>(aRow + k0 + 16 + half * 8);
      // B 32x16 frag: lane-half splits K 0-7/8-15 and 16-23/24-31 for column r15 (matches A)
      const bf16_t* bp = bRow + k0 + half * 8;
      fb.q[0] = *reinterpret_cast<const uint4*>(bp);
      fb.q[1] = *reinterpret_cast<const uint4*>(bp + 16);
      acc = __builtin_amdgcn_wmma_f32_16x16x32_bf16(false, fa.v, false, fb.v,
                                                    (short)0, acc, false, false);
    }
    const int nn = n0 + r15;
    if (nn < N) {
      float bv = bias ? bias[nn] : 0.f;
#pragma unroll
      for (int r = 0; r < 8; ++r) {
        int mm = m0 + wave * 16 + half * 8 + r;   // C layout: M = vgpr + 8*(lane>=16)
        if (mm < M) {
          float v = acc[r] + bv;
          if (accum) v += C[(size_t)mm * ldc + nn];
          C[(size_t)mm * ldc + nn] = act_apply(v, act);
        }
      }
    }
  }
}

// ---------------------------------------------------------------------------
// Utility kernels
// ---------------------------------------------------------------------------
__global__ void k_zero(float* p, int n) {
  int t = blockIdx.x * blockDim.x + threadIdx.x;
  if (t < n) p[t] = 0.f;
}
__global__ void k_copy(float* d, const float* s, int n) {
  int t = blockIdx.x * blockDim.x + threadIdx.x;
  if (t < n) d[t] = s[t];
}
__global__ void k_add(float* d, const float* a, const float* b, int n) {
  int t = blockIdx.x * blockDim.x + threadIdx.x;
  if (t < n) d[t] = a[t] + b[t];
}
__global__ void k_gather(float* __restrict__ dst, const float* __restrict__ src,
                         const int* __restrict__ idx, int L, int C) {
  int t = blockIdx.x * blockDim.x + threadIdx.x;
  if (t >= L * C) return;
  int c = t % C, l = t / C;
  dst[(size_t)l * C + c] = src[(size_t)idx[l] * C + c];
}
// f32 [N,K] -> bf16 [N16,K16] zero-padded (also used for activations, N=L)
__global__ void k_convert(const float* __restrict__ src, bf16_t* __restrict__ dst,
                          int N, int K, int N16, int K16) {
  int t = blockIdx.x * blockDim.x + threadIdx.x;
  if (t >= N16 * K16) return;
  int k = t % K16, n = t / K16;
  float v = (n < N && k < K) ? src[(size_t)n * K + k] : 0.f;
  dst[t] = (bf16_t)v;
}
// conv weight [Co,Ci,3,3,3] f32 -> tap-major bf16 [27][Co16][Ci] (contiguous K)
__global__ void k_convert_tap(const float* __restrict__ w, bf16_t* __restrict__ dst,
                              int Co, int Ci, int Co16) {
  int t = blockIdx.x * blockDim.x + threadIdx.x;
  if (t >= 27 * Co16 * Ci) return;
  int i = t % Ci;
  int o = (t / Ci) % Co16;
  int tap = t / (Ci * Co16);
  float v = (o < Co) ? w[(size_t)o * Ci * 27 + (size_t)i * 27 + tap] : 0.f;
  dst[t] = (bf16_t)v;
}
// implicit-GEMM conv: gather shifted input plane for one tap into bf16 [Lo,Ci]
// (zero rows when out of bounds -> WMMA runs with EXEC all-ones, no predication)
__global__ void k_gather_tap(const float* __restrict__ x, bf16_t* __restrict__ dst,
                             int Wi, int Hi, int Di, int Wo, int Ho, int Do,
                             int Ci, int stride, int a, int b, int c) {
  int Lo = Wo * Ho * Do;
  int t = blockIdx.x * blockDim.x + threadIdx.x;
  if (t >= Lo * Ci) return;
  int i = t % Ci, lo = t / Ci;
  int dd = lo % Do, hh = (lo / Do) % Ho, ww = lo / (Do * Ho);
  int wi = ww * stride + a - 1;
  int hi = hh * stride + b - 1;
  int di = dd * stride + c - 1;
  float v = 0.f;
  if (wi >= 0 && wi < Wi && hi >= 0 && hi < Hi && di >= 0 && di < Di)
    v = x[(size_t)((wi * Hi + hi) * Di + di) * Ci + i];
  dst[t] = (bf16_t)v;
}

// Morton permutation by rank counting (stable argsort of interleaved codes)
__global__ void k_morton(int* __restrict__ perm, int* __restrict__ inv,
                         int W, int H, int D, int nb) {
  int L = W * H * D;
  int i = blockIdx.x * blockDim.x + threadIdx.x;
  if (i >= L) return;
  auto code = [&](int l) -> unsigned {
    int d = l % D, h = (l / D) % H, w = l / (D * H);
    unsigned c = 0;
    for (int b = 0; b < nb; ++b) {
      c |= ((unsigned)((w >> b) & 1)) << (3 * b + 2);
      c |= ((unsigned)((h >> b) & 1)) << (3 * b + 1);
      c |= ((unsigned)((d >> b) & 1)) << (3 * b);
    }
    return c;
  };
  unsigned ci = code(i);
  int rank = 0;
  for (int j = 0; j < L; ++j) {
    unsigned cj = code(j);
    rank += (cj < ci) || (cj == ci && j < i);
  }
  inv[i] = rank;
  perm[rank] = i;
}

// Encoder token builder: [L,160] bf16 = concat(feats 128, coords 3, pad)
__global__ void k_build_tokens(const float* __restrict__ vf, bf16_t* __restrict__ dst,
                               int W, int H, int D) {
  int L = W * H * D;
  int t = blockIdx.x * blockDim.x + threadIdx.x;
  if (t >= L * 160) return;
  int c = t % 160, l = t / 160;
  int d = l % D, h = (l / D) % H, w = l / (D * H);
  float v = 0.f;
  if (c < 128)       v = vf[(size_t)((c * W + w) * H + h) * D + d];
  else if (c == 128) v = (float)w;
  else if (c == 129) v = (float)h;
  else if (c == 130) v = (float)d;
  dst[t] = (bf16_t)v;
}

// pos-MLP (3 -> 128 gelu -> 128) added to x; one wave per voxel (wave32)
__global__ void k_pos_add(float* __restrict__ x, const float* __restrict__ w1,
                          const float* __restrict__ b1, const float* __restrict__ w2,
                          const float* __restrict__ b2, int W, int H, int D) {
  __shared__ float sm[8 * 128];
  int L = W * H * D;
  int wave = threadIdx.x >> 5, lane = threadIdx.x & 31;
  int l = blockIdx.x * 8 + wave;
  if (l < L) {
    int d = l % D, h = (l / D) % H, w = l / (D * H);
    float cx = (float)w, cy = (float)h, cz = (float)d;
#pragma unroll
    for (int j = 0; j < 4; ++j) {
      int o = lane * 4 + j;
      float v = w1[o * 3 + 0] * cx + w1[o * 3 + 1] * cy + w1[o * 3 + 2] * cz + b1[o];
      float u = v + 0.044715f * v * v * v;
      sm[wave * 128 + o] = 0.5f * v * (1.f + tanhf(0.7978845608f * u));
    }
  }
  __syncthreads();
  if (l < L) {
#pragma unroll
    for (int j = 0; j < 4; ++j) {
      int c = lane * 4 + j;
      float acc = b2[c];
      const float* wr = w2 + (size_t)c * 128;
      const float* pr = sm + wave * 128;
      for (int k = 0; k < 128; ++k) acc += wr[k] * pr[k];
      x[(size_t)l * 128 + c] += acc;
    }
  }
}

// LayerNorm over d=128, one wave32 per token, shfl_xor reduction
__global__ void k_ln(const float* __restrict__ x, const float* __restrict__ g,
                     const float* __restrict__ b, float* outf, bf16_t* outb, int L) {
  int wave = threadIdx.x >> 5, lane = threadIdx.x & 31;
  int l = blockIdx.x * 8 + wave;
  if (l >= L) return;
  const float* row = x + (size_t)l * 128;
  float4 v = reinterpret_cast<const float4*>(row)[lane];
  float s = v.x + v.y + v.z + v.w;
  float q = v.x * v.x + v.y * v.y + v.z * v.z + v.w * v.w;
  for (int m = 1; m < 32; m <<= 1) {
    s += __shfl_xor(s, m, 32);
    q += __shfl_xor(q, m, 32);
  }
  float mean = s * (1.f / 128.f);
  float var  = q * (1.f / 128.f) - mean * mean;
  float rstd = rsqrtf(var + 1e-5f);
#pragma unroll
  for (int j = 0; j < 4; ++j) {
    int c = lane * 4 + j;
    float xv = (&v.x)[j];
    float o = (xv - mean) * rstd * g[c] + b[c];
    if (outf) outf[(size_t)l * 128 + c] = o;
    if (outb) outb[(size_t)l * 128 + c] = (bf16_t)o;
  }
}

// GroupNorm over whole spatial extent (reference reshape(B,G,-1))
__global__ void k_gn_stats(const float* __restrict__ x, float* __restrict__ sums,
                           int L, int C, int G) {
  int t = blockIdx.x * blockDim.x + threadIdx.x;
  if (t >= L * C) return;
  int c = t % C, grp = c / (C / G);
  float v = x[t];
  atomicAdd(&sums[grp * 2 + 0], v);
  atomicAdd(&sums[grp * 2 + 1], v * v);
}
__global__ void k_gn_apply(const float* __restrict__ x, float* outf, bf16_t* outb,
                           const float* __restrict__ g, const float* __restrict__ b,
                           const float* __restrict__ sums, int L, int C, int G, int relu) {
  int t = blockIdx.x * blockDim.x + threadIdx.x;
  if (t >= L * C) return;
  int c = t % C, grp = c / (C / G);
  float cnt = (float)L * (float)(C / G);
  float mean = sums[grp * 2] / cnt;
  float var = sums[grp * 2 + 1] / cnt - mean * mean;
  float v = (x[t] - mean) * rsqrtf(var + 1e-5f) * g[c] + b[c];
  if (relu && v < 0.f) v = 0.f;
  if (outf) outf[t] = v;
  if (outb) outb[t] = (bf16_t)v;
}

// Causal depthwise conv (width 4) + silu; xz layout [L,512], xh = cols 0..255
__global__ void k_dwconv_silu(const float* __restrict__ xz, const float* __restrict__ cw,
                              const float* __restrict__ cb, float* __restrict__ xcf,
                              bf16_t* __restrict__ xcb, int L) {
  int t = blockIdx.x * blockDim.x + threadIdx.x;
  if (t >= L * 256) return;
  int c = t % 256, l = t / 256;
  float acc = cb[c];
#pragma unroll
  for (int k = 0; k < 4; ++k) {
    int ls = l + k - 3;
    if (ls >= 0) acc += cw[c * 4 + k] * xz[(size_t)ls * 512 + c];
  }
  float s = acc / (1.f + __expf(-acc));
  xcf[t] = s;
  xcb[t] = (bf16_t)s;
}

// dbc[:, :8] -> bf16 [L,32] (zero-padded K for WMMA)
__global__ void k_slice_dt(const float* __restrict__ dbc, bf16_t* __restrict__ dst, int L) {
  int t = blockIdx.x * blockDim.x + threadIdx.x;
  if (t >= L * 32) return;
  int c = t % 32, l = t / 32;
  dst[t] = (bf16_t)((c < 8) ? dbc[(size_t)l * 40 + c] : 0.f);
}

// Fused selective scan: h_s <- exp(dt*A_s) h_s + dt*x*B_s ; y = h.C + x*Dp
// 256 threads = 256 channels, states in registers, never materializes [L,256,16].
__global__ void k_scan(const float* __restrict__ dt, const float* __restrict__ xc,
                       const float* __restrict__ dbc, const float* __restrict__ A_log,
                       const float* __restrict__ Dp, float* __restrict__ y, int L) {
  int d = threadIdx.x;
  float A[16], h[16];
#pragma unroll
  for (int s = 0; s < 16; ++s) { A[s] = -__expf(A_log[d * 16 + s]); h[s] = 0.f; }
  float Dpd = Dp[d];
  for (int l = 0; l < L; ++l) {
    float dtv = dt[(size_t)l * 256 + d];
    float xv = xc[(size_t)l * 256 + d];
    float dx = dtv * xv;
    const float* Br = dbc + (size_t)l * 40 + 8;
    const float* Cr = dbc + (size_t)l * 40 + 24;
    float acc = 0.f;
#pragma unroll
    for (int s = 0; s < 16; ++s) {
      h[s] = __expf(dtv * A[s]) * h[s] + dx * Br[s];
      acc += h[s] * Cr[s];
    }
    y[(size_t)l * 256 + d] = acc + xv * Dpd;
  }
}

// y * silu(z) -> bf16 [L,256]; z = xz cols 256..511
__global__ void k_gate(const float* __restrict__ y, const float* __restrict__ xz,
                       bf16_t* __restrict__ out, int L) {
  int t = blockIdx.x * blockDim.x + threadIdx.x;
  if (t >= L * 256) return;
  int c = t % 256, l = t / 256;
  float zv = xz[(size_t)l * 512 + 256 + c];
  out[t] = (bf16_t)(y[t] * (zv / (1.f + __expf(-zv))));
}

// nearest 2x upsample + channel concat with skip -> [Lo, 2C]
__global__ void k_up2_concat(const float* __restrict__ xs, const float* __restrict__ skip,
                             float* __restrict__ out, int Ws, int Hs, int Ds, int C) {
  int Wo = Ws * 2, Ho = Hs * 2, Do = Ds * 2;
  int Lo = Wo * Ho * Do;
  int t = blockIdx.x * blockDim.x + threadIdx.x;
  if (t >= Lo * C) return;
  int c = t % C, lo = t / C;
  int d = lo % Do, h = (lo / Do) % Ho, w = lo / (Do * Ho);
  int ls = ((w >> 1) * Hs + (h >> 1)) * Ds + (d >> 1);
  out[(size_t)lo * (2 * C) + c]     = xs[(size_t)ls * C + c];
  out[(size_t)lo * (2 * C) + C + c] = skip[(size_t)lo * C + c];
}

// token-major [L,17] -> channel-major [17,W,H,D]
__global__ void k_final(const float* __restrict__ h17, float* __restrict__ out, int L) {
  int t = blockIdx.x * blockDim.x + threadIdx.x;
  if (t >= 17 * L) return;
  out[t] = h17[(size_t)(t % L) * 17 + (t / L)];
}

// ---------------------------------------------------------------------------
// Host orchestration
// ---------------------------------------------------------------------------
struct MParam {
  const float *ln_g, *ln_b, *in_w, *conv_w, *conv_b, *xproj_w, *dt_w, *dt_b,
              *A_log, *Dp, *out_w;
};
struct LevelP {
  MParam lay[2];
  const float *nf_g, *nf_b, *hn_g, *hn_b;
};

extern "C" void kernel_launch(void* const* d_in, const int* in_sizes, int n_in,
                              void* d_out, int out_size, void* d_ws, size_t ws_size,
                              hipStream_t stream) {
  (void)in_sizes; (void)n_in; (void)out_size; (void)ws_size;

  // --- unpack inputs (setup_inputs insertion order) ---
  int idx = 0;
  auto nxt = [&]() -> const float* { return (const float*)d_in[idx++]; };
  const float* vfeat = nxt();
  const float* enc_w1 = nxt(); const float* enc_g1 = nxt(); const float* enc_b1 = nxt();
  const float* enc_w2 = nxt();
  const float* pos_w1 = nxt(); const float* pos_b1 = nxt();
  const float* pos_w2 = nxt(); const float* pos_b2 = nxt();
  LevelP lv[8];
  for (int i = 0; i < 8; ++i) {
    for (int j = 0; j < 2; ++j) {
      MParam& m = lv[i].lay[j];
      m.ln_g = nxt(); m.ln_b = nxt(); m.in_w = nxt(); m.conv_w = nxt(); m.conv_b = nxt();
      m.xproj_w = nxt(); m.dt_w = nxt(); m.dt_b = nxt(); m.A_log = nxt(); m.Dp = nxt();
      m.out_w = nxt();
    }
    lv[i].nf_g = nxt(); lv[i].nf_b = nxt(); lv[i].hn_g = nxt(); lv[i].hn_b = nxt();
  }
  struct CP { const float *w, *g, *b; } dnp[3], upp[3];
  for (int i = 0; i < 3; ++i) { dnp[i].w = nxt(); dnp[i].g = nxt(); dnp[i].b = nxt(); }
  for (int i = 0; i < 3; ++i) { upp[i].w = nxt(); upp[i].g = nxt(); upp[i].b = nxt(); }
  const float* occ_w1 = nxt(); const float* occ_g = nxt(); const float* occ_b = nxt();
  const float* occ_w2 = nxt();

  // --- workspace bump allocator ---
  char* wsp = (char*)d_ws; size_t off = 0;
  auto alloc = [&](size_t bytes) -> void* {
    void* p = wsp + off; off += (bytes + 255) & ~(size_t)255; return p;
  };
  const int L0 = 32 * 32 * 8;
  float* cur  = (float*)alloc((size_t)L0 * 128 * 4);
  float* res  = (float*)alloc((size_t)L0 * 128 * 4);
  float* hid  = (float*)alloc((size_t)L0 * 128 * 4);
  float* t0   = (float*)alloc((size_t)L0 * 128 * 4);
  float* seqf = (float*)alloc((size_t)L0 * 128 * 4);
  float* xz   = (float*)alloc((size_t)L0 * 512 * 4);
  float* xcf  = (float*)alloc((size_t)L0 * 256 * 4);
  float* dtb  = (float*)alloc((size_t)L0 * 256 * 4);
  float* ybuf = (float*)alloc((size_t)L0 * 256 * 4);
  float* dbc  = (float*)alloc((size_t)L0 * 40 * 4);
  float* ctmp = (float*)alloc((size_t)L0 * 128 * 4);
  float* cat  = (float*)alloc((size_t)L0 * 256 * 4);
  float* skp[3];
  skp[0] = (float*)alloc((size_t)L0 * 128 * 4);
  skp[1] = (float*)alloc((size_t)1024 * 128 * 4);
  skp[2] = (float*)alloc((size_t)128 * 128 * 4);
  float* h64 = (float*)alloc((size_t)L0 * 64 * 4);
  float* h17 = (float*)alloc((size_t)L0 * 17 * 4);
  float* gnsum = (float*)alloc(64 * 4);
  bf16_t* tokA   = (bf16_t*)alloc((size_t)L0 * 256 * 2);
  bf16_t* convA  = (bf16_t*)alloc((size_t)L0 * 256 * 2);  // implicit-GEMM gather panel
  bf16_t* xcbf   = (bf16_t*)alloc((size_t)L0 * 256 * 2);
  bf16_t* ygbf   = (bf16_t*)alloc((size_t)L0 * 256 * 2);
  bf16_t* dtinbf = (bf16_t*)alloc((size_t)L0 * 32 * 2);
  bf16_t* h64bf  = (bf16_t*)alloc((size_t)L0 * 64 * 2);

  // --- Morton permutations per distinct dims ---
  int DWs[4] = {32, 16, 8, 4}, DHs[4] = {32, 16, 8, 4}, DDs[4] = {8, 4, 2, 1};
  int* permp[4]; int* invp[4];
  for (int t = 0; t < 4; ++t) {
    int L = DWs[t] * DHs[t] * DDs[t];
    permp[t] = (int*)alloc((size_t)L * 4);
    invp[t]  = (int*)alloc((size_t)L * 4);
    int m = DWs[t]; if (DHs[t] > m) m = DHs[t]; if (DDs[t] > m) m = DDs[t];
    int nb = 0; while ((1 << nb) <= m) ++nb;
    k_morton<<<GRID(L, 256), 256, 0, stream>>>(permp[t], invp[t],
                                               DWs[t], DHs[t], DDs[t], nb);
  }

  // --- weight conversion to padded bf16 (K16 % 32, N16 % 16) ---
  auto cvt = [&](const float* src, int N, int K, int N16, int K16) -> const bf16_t* {
    bf16_t* d = (bf16_t*)alloc((size_t)N16 * K16 * 2);
    k_convert<<<GRID(N16 * K16, 256), 256, 0, stream>>>(src, d, N, K, N16, K16);
    return d;
  };
  auto cvt_tap = [&](const float* src, int Co, int Ci) -> const bf16_t* {
    bf16_t* d = (bf16_t*)alloc((size_t)27 * 128 * Ci * 2);
    k_convert_tap<<<GRID(27 * 128 * Ci, 256), 256, 0, stream>>>(src, d, Co, Ci, 128);
    return d;
  };
  const bf16_t* w_enc1 = cvt(enc_w1, 128, 131, 128, 160);
  const bf16_t* w_enc2 = cvt(enc_w2, 128, 128, 128, 128);
  const bf16_t* w_occ1 = cvt(occ_w1, 64, 128, 64, 128);
  const bf16_t* w_occ2 = cvt(occ_w2, 17, 64, 32, 64);
  const bf16_t* w_dn[3]; const bf16_t* w_up[3];
  for (int i = 0; i < 3; ++i) w_dn[i] = cvt_tap(dnp[i].w, 128, 128);
  for (int i = 0; i < 3; ++i) w_up[i] = cvt_tap(upp[i].w, 128, 256);
  struct LWbf { const bf16_t *in, *xp, *dt, *out; } lw[8][2];
  for (int i = 0; i < 8; ++i)
    for (int j = 0; j < 2; ++j) {
      lw[i][j].in  = cvt(lv[i].lay[j].in_w, 512, 128, 512, 128);
      lw[i][j].xp  = cvt(lv[i].lay[j].xproj_w, 40, 256, 48, 256);
      lw[i][j].dt  = cvt(lv[i].lay[j].dt_w, 256, 8, 256, 32);
      lw[i][j].out = cvt(lv[i].lay[j].out_w, 128, 256, 128, 256);
    }

  auto gemm = [&](const bf16_t* A, const bf16_t* Wt, float* C, const float* bias,
                  int M, int N, int K16, int N16, int ldc, int act, int accum) {
    k_gemm_wmma<<<GRID(M, TILE_M), 256, (size_t)TILE_M * K16 * 2, stream>>>(
        A, Wt, C, bias, M, N, K16, N16, ldc, act, accum);
  };
  auto gnorm = [&](const float* x, int L, int C, int G, const float* g, const float* b,
                   int relu, float* outf, bf16_t* outb) {
    k_zero<<<GRID(2 * G, 64), 64, 0, stream>>>(gnsum, 2 * G);
    k_gn_stats<<<GRID(L * C, 256), 256, 0, stream>>>(x, gnsum, L, C, G);
    k_gn_apply<<<GRID(L * C, 256), 256, 0, stream>>>(x, outf, outb, g, b, gnsum,
                                                     L, C, G, relu);
  };
  // 3x3x3 conv via implicit GEMM: 27 x (tap gather + WMMA GEMM accumulate)
  auto conv3 = [&](const float* x, const bf16_t* wt, float* out,
                   int Wi, int Hi, int Di, int Wo, int Ho, int Do, int Ci, int stride) {
    int Lo = Wo * Ho * Do;
    for (int tap = 0; tap < 27; ++tap) {
      int a = tap / 9, b = (tap / 3) % 3, c = tap % 3;
      k_gather_tap<<<GRID(Lo * Ci, 256), 256, 0, stream>>>(
          x, convA, Wi, Hi, Di, Wo, Ho, Do, Ci, stride, a, b, c);
      gemm(convA, wt + (size_t)tap * 128 * Ci, out, nullptr,
           Lo, 128, Ci, 128, 128, 0, tap > 0);
    }
  };

  // --- encoder ---
  k_build_tokens<<<GRID(L0 * 160, 256), 256, 0, stream>>>(vfeat, tokA, 32, 32, 8);
  gemm(tokA, w_enc1, t0, nullptr, L0, 128, 160, 128, 128, 0, 0);
  gnorm(t0, L0, 128, 32, enc_g1, enc_b1, 1, nullptr, tokA);
  gemm(tokA, w_enc2, cur, nullptr, L0, 128, 128, 128, 128, 0, 0);
  k_pos_add<<<GRID(L0, 8), 256, 0, stream>>>(cur, pos_w1, pos_b1, pos_w2, pos_b2,
                                             32, 32, 8);

  auto mamba = [&](int L, const MParam& mp, const LWbf& w) {
    gemm(tokA, w.in, xz, nullptr, L, 512, 128, 512, 512, 0, 0);
    k_dwconv_silu<<<GRID(L * 256, 256), 256, 0, stream>>>(xz, mp.conv_w, mp.conv_b,
                                                          xcf, xcbf, L);
    gemm(xcbf, w.xp, dbc, nullptr, L, 40, 256, 48, 40, 0, 0);
    k_slice_dt<<<GRID(L * 32, 256), 256, 0, stream>>>(dbc, dtinbf, L);
    gemm(dtinbf, w.dt, dtb, mp.dt_b, L, 256, 32, 256, 256, 3 /*softplus*/, 0);
    k_scan<<<1, 256, 0, stream>>>(dtb, xcf, dbc, mp.A_log, mp.Dp, ybuf, L);
    k_gate<<<GRID(L * 256, 256), 256, 0, stream>>>(ybuf, xz, ygbf, L);
    gemm(ygbf, w.out, hid, nullptr, L, 128, 256, 128, 128, 0, 0);
  };

  // --- 8 mixer levels with down / up path ---
  for (int i = 0; i < 8; ++i) {
    int di = (i < 4) ? i : 7 - i;
    int W = DWs[di], H = DHs[di], D = DDs[di];
    int L = W * H * D;
    // permute to Morton order
    k_gather<<<GRID(L * 128, 256), 256, 0, stream>>>(hid, cur, permp[di], L, 128);
    k_copy<<<GRID(L * 128, 256), 256, 0, stream>>>(res, hid, L * 128);
    for (int j = 0; j < 2; ++j) {
      if (j) k_add<<<GRID(L * 128, 256), 256, 0, stream>>>(res, hid, res, L * 128);
      k_ln<<<GRID(L, 8), 256, 0, stream>>>(res, lv[i].lay[j].ln_g, lv[i].lay[j].ln_b,
                                           nullptr, tokA, L);
      mamba(L, lv[i].lay[j], lw[i][j]);
    }
    k_add<<<GRID(L * 128, 256), 256, 0, stream>>>(res, hid, res, L * 128);
    k_ln<<<GRID(L, 8), 256, 0, stream>>>(res, lv[i].nf_g, lv[i].nf_b, t0, nullptr, L);
    k_ln<<<GRID(L, 8), 256, 0, stream>>>(t0, lv[i].hn_g, lv[i].hn_b, seqf, nullptr, L);
    k_gather<<<GRID(L * 128, 256), 256, 0, stream>>>(cur, seqf, invp[di], L, 128);

    if (i < 3) {  // skip + strided downconv (implicit-GEMM WMMA) + GN/relu
      k_copy<<<GRID(L * 128, 256), 256, 0, stream>>>(skp[i], cur, L * 128);
      int W2 = DWs[di + 1], H2 = DHs[di + 1], D2 = DDs[di + 1];
      int L2 = W2 * H2 * D2;
      conv3(cur, w_dn[i], ctmp, W, H, D, W2, H2, D2, 128, 2);
      gnorm(ctmp, L2, 128, 32, dnp[i].g, dnp[i].b, 1, cur, nullptr);
    }
    if (i >= 4 && i <= 6) {  // up2 + concat skip + conv (implicit-GEMM WMMA) + GN/relu
      int sb = 6 - i;
      int db = di - 1;
      int Wb = DWs[db], Hb = DHs[db], Db = DDs[db];
      int Lb = Wb * Hb * Db;
      k_up2_concat<<<GRID(Lb * 128, 256), 256, 0, stream>>>(cur, skp[sb], cat,
                                                            W, H, D, 128);
      conv3(cat, w_up[i - 4], ctmp, Wb, Hb, Db, Wb, Hb, Db, 256, 1);
      gnorm(ctmp, Lb, 128, 32, upp[i - 4].g, upp[i - 4].b, 1, cur, nullptr);
    }
  }

  // --- occupancy head ---
  k_convert<<<GRID(L0 * 128, 256), 256, 0, stream>>>(cur, tokA, L0, 128, L0, 128);
  gemm(tokA, w_occ1, h64, nullptr, L0, 64, 128, 64, 64, 0, 0);
  gnorm(h64, L0, 64, 32, occ_g, occ_b, 1, nullptr, h64bf);
  gemm(h64bf, w_occ2, h17, nullptr, L0, 17, 64, 32, 17, 0, 0);
  k_final<<<GRID(L0 * 17, 256), 256, 0, stream>>>(h17, (float*)d_out, L0);
}
